// CVRPActorPairs_89249420411560
// MI455X (gfx1250) — matrix operand order repaired and checked by reading, not compiled, not run on hardware
//
#include <hip/hip_runtime.h>
#include <math.h>

// ---------------------------------------------------------------------------
// CVRP actor pairs MLP for MI455X (gfx1250, wave32, WMMA 16x16x32 f16->f32)
//
//   B=16, N=512, EMBED=32, C_IN=20, P = N*(N-1)/2 = 130816 = 256 * 511
//   8 waves/block, TWO 16-pair tiles per wave (weight fragments + biases
//   amortized across both), 511 blocks per batch element.
//
// WMMA operand roles (all LDS traffic is b128):
//   A = W^T tile (16 out-channels x 32 in-channels), staged in LDS row-major
//   B = activations^T (32 in-channels x 16 pairs), stored pair-major [pair][ch]
//   D = h^T (out-channel x pair) -> packed f16 relu -> b128 store pair-major
// ---------------------------------------------------------------------------

typedef __attribute__((ext_vector_type(16))) _Float16 v16h;
typedef __attribute__((ext_vector_type(8)))  _Float16 v8h;
typedef __attribute__((ext_vector_type(8)))  float    v8f;

#define B_               16
#define N_               512
#define P_               130816
#define EMBED_           32
#define CS_STRIDE        12     // 9 c_state feats + temp + time + pad (48 B rows)
#define WAVES_PER_BLOCK  8
#define TILES_PER_WAVE   2
#define PAIRS_PER_BLOCK  (WAVES_PER_BLOCK * TILES_PER_WAVE * 16)   // 256
#define BLOCKS_PER_BATCH (P_ / PAIRS_PER_BLOCK)                    // 511

// ---------------------------------------------------------------------------
// Kernel 1: per-node feature table
//   cs[b][n] = [coords@x(n) (2), coords@x(n-1) (2), coords@x(n+1) (2),
//               extra (3), temp, time, pad]
// ---------------------------------------------------------------------------
__global__ __launch_bounds__(256)
void cvrp_prep_kernel(const float* __restrict__ state, float* __restrict__ cs)
{
    int t = blockIdx.x * blockDim.x + threadIdx.x;
    if (t >= B_ * N_) return;
    int b = t / N_, n = t % N_;
    const float* base = state + (size_t)b * N_ * 8;

    int np = (n + N_ - 1) & (N_ - 1);
    int nx = (n + 1) & (N_ - 1);
    int xn = (int)base[(size_t)n  * 8];
    int xp = (int)base[(size_t)np * 8];
    int xx = (int)base[(size_t)nx * 8];
    xn = min(max(xn, 0), N_ - 1);
    xp = min(max(xp, 0), N_ - 1);
    xx = min(max(xx, 0), N_ - 1);

    float* o = cs + (size_t)t * CS_STRIDE;
    o[0]  = base[(size_t)xn * 8 + 1];
    o[1]  = base[(size_t)xn * 8 + 2];
    o[2]  = base[(size_t)xp * 8 + 1];
    o[3]  = base[(size_t)xp * 8 + 2];
    o[4]  = base[(size_t)xx * 8 + 1];
    o[5]  = base[(size_t)xx * 8 + 2];
    o[6]  = base[(size_t)n  * 8 + 3];
    o[7]  = base[(size_t)n  * 8 + 4];
    o[8]  = base[(size_t)n  * 8 + 5];
    o[9]  = base[(size_t)n  * 8 + 6];   // temp
    o[10] = base[(size_t)n  * 8 + 7];   // time
    o[11] = 0.0f;
}

// ---------------------------------------------------------------------------
// Layout helpers (gfx1250 wave32, CDNA5 ISA §7.12.2)
//   m = lane%16, h = lane/16
//   A 16x32 f16: lane row = m (+16 for tile1); elems 0..7  = K 8h..8h+7,
//                                              elems 8..15 = K 16+8h..16+8h+7
//   B 32x16 f16: lane col = m; elems 0..7 = K 8h..8h+7, 8..15 = K 16+8h..16+8h+7
//   C/D 16x16 f32: lane col = m; VGPR r = row r+8h
// ---------------------------------------------------------------------------
__device__ inline v16h load_a_wt(const _Float16* wT, int row, int h)
{
    const v8h* wr = (const v8h*)(wT + row * EMBED_);
    v8h lo = wr[h];        // K = 8h .. 8h+7
    v8h hi = wr[2 + h];    // K = 16+8h .. 16+8h+7
    return __builtin_shufflevector(lo, hi, 0, 1, 2, 3, 4, 5, 6, 7,
                                           8, 9, 10, 11, 12, 13, 14, 15);
}

__device__ inline v16h load_b_act(const _Float16* bt, int m, int h)
{
    const v8h* br = (const v8h*)(bt + m * EMBED_);
    v8h lo = br[h];        // K = 8h .. 8h+7
    v8h hi = br[2 + h];    // K = 16+8h .. 16+8h+7
    return __builtin_shufflevector(lo, hi, 0, 1, 2, 3, 4, 5, 6, 7,
                                           8, 9, 10, 11, 12, 13, 14, 15);
}

// Pack to f16 first (v_cvt_pk_f16_f32, needed anyway), then relu as packed
// v_pk_max_num_f16 -- avoids the 2-op canonicalizing f32 fmaxf sequence.
__device__ inline v8h relu_pack(v8f d)
{
    v8h p;
#pragma unroll
    for (int i = 0; i < 8; ++i) p[i] = (_Float16)d[i];
    return __builtin_elementwise_max(p, (v8h)(_Float16)0.0f);
}

// Single-instruction f32 relu: v_med3(x, 0, +inf)
__device__ inline float relu1(float x)
{
    return __builtin_amdgcn_fmed3f(x, 0.0f, __builtin_inff());
}

// One layer over TWO pair-tiles: h^T = relu(W^T * act^T + bias) for each.
// Weight fragments a0/a1 and bias registers are shared across both tiles.
__device__ inline void mlp_layer2(_Float16* bt0, _Float16* bt1,
                                  const _Float16* wT,
                                  const float* __restrict__ bias,
                                  int m, int h)
{
    v16h a0  = load_a_wt(wT, m,      h);
    v16h a1  = load_a_wt(wT, m + 16, h);
    v16h bv0 = load_b_act(bt0, m, h);
    v16h bv1 = load_b_act(bt1, m, h);

    v8f bia0, bia1;
#pragma unroll
    for (int r = 0; r < 8; ++r) {
        bia0[r] = bias[r + 8 * h];
        bia1[r] = bias[16 + r + 8 * h];
    }

    v8f d00 = __builtin_amdgcn_wmma_f32_16x16x32_f16(
        false, a0, false, bv0, (short)0, bia0, false, false);
    v8f d01 = __builtin_amdgcn_wmma_f32_16x16x32_f16(
        false, a1, false, bv0, (short)0, bia1, false, false);
    v8f d10 = __builtin_amdgcn_wmma_f32_16x16x32_f16(
        false, a0, false, bv1, (short)0, bia0, false, false);
    v8f d11 = __builtin_amdgcn_wmma_f32_16x16x32_f16(
        false, a1, false, bv1, (short)0, bia1, false, false);

    v8h* row0 = (v8h*)(bt0 + m * EMBED_);
    v8h* row1 = (v8h*)(bt1 + m * EMBED_);
    row0[h]     = relu_pack(d00);
    row0[2 + h] = relu_pack(d01);
    row1[h]     = relu_pack(d10);
    row1[2 + h] = relu_pack(d11);
}

// ---------------------------------------------------------------------------
// Kernel 2: gather pair features, 3 WMMA layers, shuffle-reduced final dot
// ---------------------------------------------------------------------------
__global__ __launch_bounds__(256)
void cvrp_mlp_kernel(const float* __restrict__ cs,
                     const int* __restrict__ idx1,
                     const int* __restrict__ idx2,
                     const unsigned char* __restrict__ mask,
                     const float* __restrict__ W1, const float* __restrict__ b1,
                     const float* __restrict__ W2, const float* __restrict__ b2,
                     const float* __restrict__ W3, const float* __restrict__ b3,
                     const float* __restrict__ W4,
                     float* __restrict__ out)
{
    // W^T tiles in LDS; layer-1 rows permuted so the gather is 2x b128/lane:
    //   k_new 0..8  = p1 feats (W1 rows 0..8)     k_new 9  = temp (row 18)
    //   k_new 10    = time (row 19)               k_new 11..15 = zero pad
    //   k_new 16..24= p2 feats (W1 rows 9..17)    k_new 25..31 = zero pad
    __shared__ __align__(16) _Float16 wlds[3][EMBED_ * EMBED_];
    __shared__ __align__(16) _Float16 btile[WAVES_PER_BLOCK][TILES_PER_WAVE][16 * EMBED_];

    const int tid = threadIdx.x;

    for (int i = tid; i < 3 * EMBED_ * EMBED_; i += 256) {
        int mat = i >> 10, rem = i & 1023, outc = rem >> 5, in = rem & 31;
        float v;
        if (mat == 0) {
            int src;
            if (in < 9)                   src = in;        // p1 feats
            else if (in == 9)             src = 18;        // temp
            else if (in == 10)            src = 19;        // time
            else if (in >= 16 && in < 25) src = in - 7;    // p2 feats
            else                          src = -1;        // zero pad
            v = (src >= 0) ? W1[src * EMBED_ + outc] : 0.0f;
        } else if (mat == 1) {
            v = W2[in * EMBED_ + outc];
        } else {
            v = W3[in * EMBED_ + outc];
        }
        wlds[mat][rem] = (_Float16)v;
    }

    const int wave = tid >> 5;
    const int lane = tid & 31;
    const int m    = lane & 15;
    const int h    = lane >> 4;
    const int b    = blockIdx.x / BLOCKS_PER_BATCH;
    const int blk  = blockIdx.x % BLOCKS_PER_BATCH;
    const int p0   = blk * PAIRS_PER_BLOCK + wave * (TILES_PER_WAVE * 16);

    // Gather both tiles: half0 writes k 0..15 (p1 feats + temp + time + pad),
    //                    half1 writes k 16..31 (p2 feats + pad).
    _Float16* bt0 = btile[wave][0];
    _Float16* bt1 = btile[wave][1];
#pragma unroll
    for (int t = 0; t < TILES_PER_WAVE; ++t) {
        const int p    = p0 + 16 * t + m;
        const int node = h ? idx2[p] : idx1[p];
        const float4* r = (const float4*)(cs + ((size_t)b * N_ + node) * CS_STRIDE);
        float4 a0 = r[0];            // feats 0..3
        float4 a1 = r[1];            // feats 4..7
        float4 a2 = r[2];            // feat 8, temp, time, pad

        v8h c0, c1;
        c0[0] = (_Float16)a0.x; c0[1] = (_Float16)a0.y;
        c0[2] = (_Float16)a0.z; c0[3] = (_Float16)a0.w;
        c0[4] = (_Float16)a1.x; c0[5] = (_Float16)a1.y;
        c0[6] = (_Float16)a1.z; c0[7] = (_Float16)a1.w;
        c1 = (v8h)(_Float16)0.0f;
        c1[0] = (_Float16)a2.x;                 // feat 8
        if (!h) {
            c1[1] = (_Float16)a2.y;             // temp (at idx1)
            c1[2] = (_Float16)a2.z;             // time (at idx1)
        }
        v8h* row = (v8h*)(btile[wave][t] + m * EMBED_);
        row[2 * h]     = c0;
        row[2 * h + 1] = c1;
    }
    __syncthreads();   // weights staged + gather visible

    mlp_layer2(bt0, bt1, wlds[0], b1, m, h);
    __syncthreads();
    mlp_layer2(bt0, bt1, wlds[1], b2, m, h);
    __syncthreads();

    // Layer 3 fused with layer 4: keep D in registers, dot with W4,
    // cross-half shuffle-add (lanes m and m+16 hold complementary channels).
    {
        v16h a0  = load_a_wt(wlds[2], m,      h);
        v16h a1  = load_a_wt(wlds[2], m + 16, h);
        v16h bv0 = load_b_act(bt0, m, h);
        v16h bv1 = load_b_act(bt1, m, h);

        v8f bia0, bia1, w40, w41;
#pragma unroll
        for (int r = 0; r < 8; ++r) {
            bia0[r] = b3[r + 8 * h];
            bia1[r] = b3[16 + r + 8 * h];
            w40[r]  = W4[r + 8 * h];
            w41[r]  = W4[16 + r + 8 * h];
        }

        v8f d00 = __builtin_amdgcn_wmma_f32_16x16x32_f16(
            false, a0, false, bv0, (short)0, bia0, false, false);
        v8f d01 = __builtin_amdgcn_wmma_f32_16x16x32_f16(
            false, a1, false, bv0, (short)0, bia1, false, false);
        v8f d10 = __builtin_amdgcn_wmma_f32_16x16x32_f16(
            false, a0, false, bv1, (short)0, bia0, false, false);
        v8f d11 = __builtin_amdgcn_wmma_f32_16x16x32_f16(
            false, a1, false, bv1, (short)0, bia1, false, false);

        float part0 = 0.0f, part1 = 0.0f;
#pragma unroll
        for (int r = 0; r < 8; ++r) {
            part0 += relu1(d00[r]) * w40[r];
            part0 += relu1(d01[r]) * w41[r];
            part1 += relu1(d10[r]) * w40[r];
            part1 += relu1(d11[r]) * w41[r];
        }
        float logit0 = part0 + __shfl_xor(part0, 16, 32);
        float logit1 = part1 + __shfl_xor(part1, 16, 32);

        if (h == 0) {
            size_t o0 = (size_t)b * P_ + (size_t)(p0 + m);
            size_t o1 = o0 + 16;
            out[o0] = mask[o0] ? -__builtin_inff() : logit0;
            out[o1] = mask[o1] ? -__builtin_inff() : logit1;
        }
    }
}

// ---------------------------------------------------------------------------
// Kernel 3: in-place log_softmax over each batch row of P_ logits
// ---------------------------------------------------------------------------
__global__ __launch_bounds__(1024)
void cvrp_lsm_kernel(float* __restrict__ out)
{
    __shared__ float red[1024];
    const int b   = blockIdx.x;
    const int tid = threadIdx.x;
    float* row = out + (size_t)b * P_;
    const float NEG_INF = -__builtin_inff();

    float lmax = NEG_INF;
    for (int i = tid; i < P_; i += 1024) lmax = fmaxf(lmax, row[i]);
    red[tid] = lmax;
    __syncthreads();
    for (int s = 512; s > 0; s >>= 1) {
        if (tid < s) red[tid] = fmaxf(red[tid], red[tid + s]);
        __syncthreads();
    }
    float gmax = red[0];
    __syncthreads();

    float lsum = 0.0f;
    for (int i = tid; i < P_; i += 1024) {
        float v = row[i];
        if (v != NEG_INF) lsum += __expf(v - gmax);
    }
    red[tid] = lsum;
    __syncthreads();
    for (int s = 512; s > 0; s >>= 1) {
        if (tid < s) red[tid] += red[tid + s];
        __syncthreads();
    }
    float lse = __logf(red[0]);

    for (int i = tid; i < P_; i += 1024) {
        float v = row[i];
        row[i] = v - gmax - lse;
    }
}

// ---------------------------------------------------------------------------
extern "C" void kernel_launch(void* const* d_in, const int* in_sizes, int n_in,
                              void* d_out, int out_size, void* d_ws, size_t ws_size,
                              hipStream_t stream)
{
    (void)in_sizes; (void)n_in; (void)out_size; (void)ws_size;

    const float*         state = (const float*)d_in[0];
    const unsigned char* mask  = (const unsigned char*)d_in[1];
    const int*           idx1  = (const int*)d_in[2];
    const int*           idx2  = (const int*)d_in[3];
    const float*         W1    = (const float*)d_in[4];
    const float*         b1    = (const float*)d_in[5];
    const float*         W2    = (const float*)d_in[6];
    const float*         b2    = (const float*)d_in[7];
    const float*         W3    = (const float*)d_in[8];
    const float*         b3    = (const float*)d_in[9];
    const float*         W4    = (const float*)d_in[10];
    float*               out   = (float*)d_out;
    float*               cs    = (float*)d_ws;   // B*N*12 floats = 384 KiB

    cvrp_prep_kernel<<<(B_ * N_ + 255) / 256, 256, 0, stream>>>(state, cs);
    cvrp_mlp_kernel<<<B_ * BLOCKS_PER_BATCH, 256, 0, stream>>>(
        cs, idx1, idx2, mask, W1, b1, W2, b2, W3, b3, W4, out);
    cvrp_lsm_kernel<<<B_, 1024, 0, stream>>>(out);
}
